// RoPEAttention_13426067767774
// MI455X (gfx1250) — matrix operand order repaired and checked by reading, not compiled, not run on hardware
//
#include <hip/hip_runtime.h>
#include <math.h>
#include <stdint.h>

// RoPE multi-head attention forward for MI455X (gfx1250, wave32, WMMA).
// B=2, S=2048, DIM=1024, H=16, HD=64. All GEMMs in bf16 WMMA, fp32 accum.
// Data movement: GLOBAL_LOAD_ASYNC_TO_LDS (ASYNCcnt) + double-buffered LDS.

#define BB   2
#define SEQ  2048
#define DIMV 1024
#define NH   16
#define HD   64

typedef __attribute__((ext_vector_type(16))) __bf16 v16bf;
typedef __attribute__((ext_vector_type(8)))  __bf16 v8bf;
typedef __attribute__((ext_vector_type(8)))  float  v8f;
typedef __attribute__((ext_vector_type(4)))  int    v4i;

// ---------------------------------------------------------------------------
// CDNA5 async copy (global -> LDS), guarded so the file still compiles if the
// toolchain lacks the builtin. Builtin signature (from toolchain diagnostic):
//   void __builtin_amdgcn_global_load_async_to_lds_b128(
//       v4i addrspace(1)* src, v4i addrspace(3)* dst, imm int off, imm int cpol)
// ---------------------------------------------------------------------------
#if defined(__has_builtin)
#  if __has_builtin(__builtin_amdgcn_global_load_async_to_lds_b128)
#    define USE_ASYNC 1
#  endif
#endif
#ifndef USE_ASYNC
#  define USE_ASYNC 0
#endif

#if USE_ASYNC
#  if __has_builtin(__builtin_amdgcn_s_wait_asynccnt)
#    define WAIT_ASYNC() __builtin_amdgcn_s_wait_asynccnt(0)
#  else
#    define WAIT_ASYNC() asm volatile("s_wait_asynccnt 0x0" ::: "memory")
#  endif
#else
#  define WAIT_ASYNC() ((void)0)
#endif

static __device__ __forceinline__ v8bf load8(const __bf16* p) {
  return *(const v8bf*)p;
}

// Copy 16 bytes (8 bf16) from global to LDS. Async path uses
// global_load_async_to_lds_b128; LDS offset = low 32 bits of the generic
// pointer (ISA: LDS aperture address truncates to addr[31:0]).
static __device__ __forceinline__ void cp16(const __bf16* g, __bf16* l) {
#if USE_ASYNC
  __builtin_amdgcn_global_load_async_to_lds_b128(
      (__attribute__((address_space(1))) v4i*)(uintptr_t)g,
      (__attribute__((address_space(3))) v4i*)(uint32_t)(uintptr_t)l,
      0, 0);
#else
  *(v8bf*)l = *(const v8bf*)g;
#endif
}

static __device__ __forceinline__ v8f wmma_bf16(v16bf a, v16bf b, v8f c) {
  return __builtin_amdgcn_wmma_f32_16x16x32_bf16(
      false, a, false, b, (short)0, c, false, false);
}

// A fragment (16x32, 16-bit): lane l holds row m=l&15;
// elems 0..7 -> K = hi*8+j, elems 8..15 -> K = 16+hi*8+j  (hi = l>>4).
static __device__ __forceinline__ v16bf frag_a(const __bf16* base, int m, int hi, int rowstride) {
  const __bf16* r = base + (size_t)m * rowstride;
  v8bf lo = load8(r + hi * 8);
  v8bf up = load8(r + 16 + hi * 8);
  v16bf a;
#pragma unroll
  for (int j = 0; j < 8; ++j) { a[j] = lo[j]; a[j + 8] = up[j]; }
  return a;
}

// B fragment (32x16, 16-bit): lane l holds column n=l&15;
// elems 0..7 -> K = hi*8+j, elems 8..15 -> K = 16+hi*8+j. Element (k,n) at base[n*rowstride + k].
static __device__ __forceinline__ v16bf frag_b(const __bf16* base, int n, int hi, int rowstride) {
  const __bf16* r = base + (size_t)n * rowstride;
  v8bf lo = load8(r + hi * 8);
  v8bf up = load8(r + 16 + hi * 8);
  v16bf b;
#pragma unroll
  for (int j = 0; j < 8; ++j) { b[j] = lo[j]; b[j + 8] = up[j]; }
  return b;
}

// ---------------------------------------------------------------------------
// Kernel 0: fp32 -> bf16 conversion (vectorized, n % 8 == 0)
// ---------------------------------------------------------------------------
__global__ void cvt_kernel(const float* __restrict__ s, __bf16* __restrict__ d, int n) {
  int i = (blockIdx.x * blockDim.x + threadIdx.x) * 8;
  int stride = gridDim.x * blockDim.x * 8;
  for (; i < n; i += stride) {
    float4 a = *(const float4*)(s + i);
    float4 b = *(const float4*)(s + i + 4);
    v8bf o;
    o[0] = (__bf16)a.x; o[1] = (__bf16)a.y; o[2] = (__bf16)a.z; o[3] = (__bf16)a.w;
    o[4] = (__bf16)b.x; o[5] = (__bf16)b.y; o[6] = (__bf16)b.z; o[7] = (__bf16)b.w;
    *(v8bf*)(d + i) = o;
  }
}

// ---------------------------------------------------------------------------
// Kernel 1: Q/K/V projection + fused RoPE.
// grid = (BB*SEQ/128, NH, 3); block = 256 (8 waves).
// Double-buffered 64-deep K tiles loaded via async-to-LDS.
// ---------------------------------------------------------------------------
__global__ __launch_bounds__(256)
void qkv_rope_kernel(const __bf16* __restrict__ xb,
                     const __bf16* __restrict__ wqb,
                     const __bf16* __restrict__ wkb,
                     const __bf16* __restrict__ wvb,
                     __bf16* __restrict__ qo,
                     __bf16* __restrict__ ko,
                     __bf16* __restrict__ vo) {
  __shared__ __bf16 xs[2][128][64];  // 32 KB
  __shared__ __bf16 wt[2][64][64];   // 16 KB
  const int t = threadIdx.x;
  const int sel = blockIdx.z;
  const int h = blockIdx.y;
  const int m0 = blockIdx.x * 128;
  const __bf16* w = (sel == 0) ? wqb : ((sel == 1) ? wkb : wvb);
  __bf16* out = (sel == 0) ? qo : ((sel == 1) ? ko : vo);

  const int wave = t >> 5, lane = t & 31;
  const int hi = lane >> 4, ln = lane & 15;

  auto load_tiles = [&](int kk, int bf) {
    int row = t >> 1, ch = (t & 1) * 32;           // 2 threads/row, 32 elems each
    const __bf16* src = xb + (size_t)(m0 + row) * DIMV + kk + ch;
    cp16(src,      &xs[bf][row][ch]);
    cp16(src + 8,  &xs[bf][row][ch + 8]);
    cp16(src + 16, &xs[bf][row][ch + 16]);
    cp16(src + 24, &xs[bf][row][ch + 24]);
    int wr = t >> 2, wc = (t & 3) * 16;            // 4 threads/row, 16 elems each
    const __bf16* ws = w + (size_t)(h * HD + wr) * DIMV + kk + wc;
    cp16(ws,     &wt[bf][wr][wc]);
    cp16(ws + 8, &wt[bf][wr][wc + 8]);
  };

  load_tiles(0, 0);
  WAIT_ASYNC();
  __syncthreads();

  v8f acc[4] = {};
  for (int kk = 0; kk < DIMV; kk += 64) {
    int cur = (kk >> 6) & 1;
    if (kk + 64 < DIMV) load_tiles(kk + 64, cur ^ 1);

    v16bf a0 = frag_a(&xs[cur][wave * 16][0], ln, hi, 64);
    v16bf a1 = frag_a(&xs[cur][wave * 16][32], ln, hi, 64);
    v16bf b0[4], b1[4];
#pragma unroll
    for (int u = 0; u < 4; ++u) {
      b0[u] = frag_b(&wt[cur][u * 16][0], ln, hi, 64);
      b1[u] = frag_b(&wt[cur][u * 16][32], ln, hi, 64);
    }
#pragma unroll
    for (int u = 0; u < 4; ++u) acc[u] = wmma_bf16(a0, b0[u], acc[u]);
#pragma unroll
    for (int u = 0; u < 4; ++u) acc[u] = wmma_bf16(a1, b1[u], acc[u]);

    WAIT_ASYNC();
    __syncthreads();
  }

  // Fused RoPE: pair (d, d+32) sits in acc[u] / acc[u+2], same lane & VGPR.
  if (sel < 2) {
    const float NEG = -0.287823137f;  // -ln(10000)/32
#pragma unroll
    for (int u = 0; u < 2; ++u) {
      float fr = __expf((float)(u * 16 + ln) * NEG);
#pragma unroll
      for (int r = 0; r < 8; ++r) {
        int g = m0 + wave * 16 + r + 8 * hi;
        float ang = (float)(g & (SEQ - 1)) * fr;
        float sa, ca;
        __sincosf(ang, &sa, &ca);
        float x1 = acc[u][r], x2 = acc[u + 2][r];
        acc[u][r]     = x1 * ca - x2 * sa;
        acc[u + 2][r] = x1 * sa + x2 * ca;
      }
    }
  }

  // Store bf16 [B, H, S, HD]
#pragma unroll
  for (int u = 0; u < 4; ++u)
#pragma unroll
    for (int r = 0; r < 8; ++r) {
      int g = m0 + wave * 16 + r + 8 * hi;
      int b = g >> 11;             // g / SEQ
      int s = g & (SEQ - 1);
      out[((size_t)(b * NH + h) * SEQ + s) * HD + u * 16 + ln] = (__bf16)acc[u][r];
    }
}

// ---------------------------------------------------------------------------
// Kernel 2: flash attention. grid = (SEQ/128, NH, BB); block = 256.
// 8 waves x 16 queries; K tiles double-buffered via async-to-LDS,
// V tiles transposed-on-write (double-buffered), P staged per-wave in LDS.
// ---------------------------------------------------------------------------
__global__ __launch_bounds__(256)
void attn_kernel(const __bf16* __restrict__ q, const __bf16* __restrict__ k,
                 const __bf16* __restrict__ v, __bf16* __restrict__ o) {
  __shared__ __bf16 ks[2][32][64];    // K tile: [key][dim]        8 KB
  __shared__ __bf16 vt[2][64][40];    // V transposed, padded      10 KB
  __shared__ __bf16 ps[8][16][40];    // per-wave P staging        10 KB

  const int t = threadIdx.x, wave = t >> 5, lane = t & 31;
  const int hi = lane >> 4, ln = lane & 15;
  const int h = blockIdx.y, b = blockIdx.z;
  const int q0 = blockIdx.x * 128 + wave * 16;
  const size_t hb = (size_t)(b * NH + h) * SEQ * HD;
  const float scale = 0.125f;  // 64^-0.5

  const int key = t >> 3, cb = (t & 7) * 8;  // cooperative tile mapping

  auto load_k = [&](int kt, int bf) {
    cp16(k + hb + (size_t)(kt + key) * HD + cb, &ks[bf][key][cb]);
  };
  auto load_v = [&](int kt) {
    return load8(v + hb + (size_t)(kt + key) * HD + cb);
  };
  auto store_v = [&](v8bf vv, int bf) {
#pragma unroll
    for (int j = 0; j < 8; ++j) vt[bf][cb + j][key] = vv[j];
  };

  // Q fragments for this wave's 16 query rows (two 16x32 A frags)
  v16bf aq0 = frag_a(q + hb + (size_t)q0 * HD, ln, hi, HD);
  v16bf aq1 = frag_a(q + hb + (size_t)q0 * HD + 32, ln, hi, HD);

  float mrow[8], lrow[8];
#pragma unroll
  for (int r = 0; r < 8; ++r) { mrow[r] = -INFINITY; lrow[r] = 0.0f; }
  v8f oacc[4] = {};

  // prologue: stage tile 0
  load_k(0, 0);
  store_v(load_v(0), 0);
  WAIT_ASYNC();
  __syncthreads();

  for (int kt = 0; kt < SEQ; kt += 32) {
    const int cur = (kt >> 5) & 1;
    const bool more = (kt + 32 < SEQ);
    v8bf vnext;
    if (more) {            // issue next tile early (async K + global V)
      load_k(kt + 32, cur ^ 1);
      vnext = load_v(kt + 32);
    }

    // scores S = Q K^T  (two 16-key column tiles, contraction over hd=64)
    v16bf bk[4];
    bk[0] = frag_b(&ks[cur][0][0],   ln, hi, 64);
    bk[1] = frag_b(&ks[cur][0][32],  ln, hi, 64);
    bk[2] = frag_b(&ks[cur][16][0],  ln, hi, 64);
    bk[3] = frag_b(&ks[cur][16][32], ln, hi, 64);
    v8f sc[2] = {};
    sc[0] = wmma_bf16(aq0, bk[0], sc[0]);
    sc[0] = wmma_bf16(aq1, bk[1], sc[0]);
    sc[1] = wmma_bf16(aq0, bk[2], sc[1]);
    sc[1] = wmma_bf16(aq1, bk[3], sc[1]);

    // online softmax; C-layout row = r + 8*hi, col = ln
    float pl[2][8];
#pragma unroll
    for (int r = 0; r < 8; ++r) {
      float v0 = sc[0][r] * scale, v1 = sc[1][r] * scale;
      float mx = fmaxf(v0, v1);
#pragma unroll
      for (int m = 1; m < 16; m <<= 1) mx = fmaxf(mx, __shfl_xor(mx, m, 32));
      float mnew = fmaxf(mrow[r], mx);
      float al = __expf(mrow[r] - mnew);
      float p0 = __expf(v0 - mnew), p1 = __expf(v1 - mnew);
      float psum = p0 + p1;
#pragma unroll
      for (int m = 1; m < 16; m <<= 1) psum += __shfl_xor(psum, m, 32);
      lrow[r] = lrow[r] * al + psum;
      mrow[r] = mnew;
      pl[0][r] = p0; pl[1][r] = p1;
#pragma unroll
      for (int u = 0; u < 4; ++u) oacc[u][r] *= al;
    }

    // P: C layout -> row-major LDS (wave-private), reload as A fragment
#pragma unroll
    for (int u = 0; u < 2; ++u)
#pragma unroll
      for (int r = 0; r < 8; ++r)
        ps[wave][r + 8 * hi][u * 16 + ln] = (__bf16)pl[u][r];
    asm volatile("s_wait_dscnt 0x0" ::: "memory");
    v16bf ap = frag_a(&ps[wave][0][0], ln, hi, 40);

    // O += P V  (4 dim tiles of 16, contraction over 32 keys)
    v16bf bv[4];
#pragma unroll
    for (int u = 0; u < 4; ++u) bv[u] = frag_b(&vt[cur][u * 16][0], ln, hi, 40);
#pragma unroll
    for (int u = 0; u < 4; ++u) oacc[u] = wmma_bf16(ap, bv[u], oacc[u]);

    if (more) store_v(vnext, cur ^ 1);  // prev readers of buf cur^1 done last iter
    WAIT_ASYNC();
    __syncthreads();
  }

  // normalize and store bf16 [B, S, DIM] (dim = h*64 + d)
#pragma unroll
  for (int u = 0; u < 4; ++u)
#pragma unroll
    for (int r = 0; r < 8; ++r) {
      int row = q0 + r + 8 * hi;
      float val = oacc[u][r] / lrow[r];
      o[((size_t)b * SEQ + row) * DIMV + h * HD + u * 16 + ln] = (__bf16)val;
    }
}

// ---------------------------------------------------------------------------
// Kernel 3: output projection + bias, fp32 out. grid = (BB*SEQ/128, DIMV/64).
// ---------------------------------------------------------------------------
__global__ __launch_bounds__(256)
void oproj_kernel(const __bf16* __restrict__ ob, const __bf16* __restrict__ wob,
                  const float* __restrict__ bias, float* __restrict__ out) {
  __shared__ __bf16 xs[2][128][64];
  __shared__ __bf16 wt[2][64][64];
  const int t = threadIdx.x, wave = t >> 5, lane = t & 31;
  const int hi = lane >> 4, ln = lane & 15;
  const int m0 = blockIdx.x * 128, n0 = blockIdx.y * 64;

  auto load_tiles = [&](int kk, int bf) {
    int row = t >> 1, ch = (t & 1) * 32;
    const __bf16* src = ob + (size_t)(m0 + row) * DIMV + kk + ch;
    cp16(src,      &xs[bf][row][ch]);
    cp16(src + 8,  &xs[bf][row][ch + 8]);
    cp16(src + 16, &xs[bf][row][ch + 16]);
    cp16(src + 24, &xs[bf][row][ch + 24]);
    int wr = t >> 2, wc = (t & 3) * 16;
    const __bf16* ws = wob + (size_t)(n0 + wr) * DIMV + kk + wc;
    cp16(ws,     &wt[bf][wr][wc]);
    cp16(ws + 8, &wt[bf][wr][wc + 8]);
  };

  load_tiles(0, 0);
  WAIT_ASYNC();
  __syncthreads();

  v8f acc[4] = {};
  for (int kk = 0; kk < DIMV; kk += 64) {
    int cur = (kk >> 6) & 1;
    if (kk + 64 < DIMV) load_tiles(kk + 64, cur ^ 1);

    v16bf a0 = frag_a(&xs[cur][wave * 16][0], ln, hi, 64);
    v16bf a1 = frag_a(&xs[cur][wave * 16][32], ln, hi, 64);
    v16bf b0[4], b1[4];
#pragma unroll
    for (int u = 0; u < 4; ++u) {
      b0[u] = frag_b(&wt[cur][u * 16][0], ln, hi, 64);
      b1[u] = frag_b(&wt[cur][u * 16][32], ln, hi, 64);
    }
#pragma unroll
    for (int u = 0; u < 4; ++u) acc[u] = wmma_bf16(a0, b0[u], acc[u]);
#pragma unroll
    for (int u = 0; u < 4; ++u) acc[u] = wmma_bf16(a1, b1[u], acc[u]);

    WAIT_ASYNC();
    __syncthreads();
  }

#pragma unroll
  for (int u = 0; u < 4; ++u) {
    float bv = bias[n0 + u * 16 + ln];
#pragma unroll
    for (int r = 0; r < 8; ++r) {
      int g = m0 + wave * 16 + r + 8 * hi;
      out[(size_t)g * DIMV + n0 + u * 16 + ln] = acc[u][r] + bv;
    }
  }
}

// ---------------------------------------------------------------------------
// Launch
// ---------------------------------------------------------------------------
extern "C" void kernel_launch(void* const* d_in, const int* in_sizes, int n_in,
                              void* d_out, int out_size, void* d_ws, size_t ws_size,
                              hipStream_t stream) {
  (void)in_sizes; (void)n_in; (void)out_size; (void)ws_size;
  const float* x   = (const float*)d_in[0];
  const float* wq  = (const float*)d_in[1];
  const float* wk  = (const float*)d_in[2];
  const float* wv  = (const float*)d_in[3];
  const float* wo  = (const float*)d_in[4];
  const float* wob = (const float*)d_in[5];

  char* wsp = (char*)d_ws;
  const size_t MB = 1024ull * 1024ull;
  __bf16* xb   = (__bf16*)(wsp);             //  8 MiB  x bf16 [4096,1024]
  __bf16* wqb  = (__bf16*)(wsp +  8 * MB);   //  2 MiB
  __bf16* wkb  = (__bf16*)(wsp + 10 * MB);   //  2 MiB
  __bf16* wvb  = (__bf16*)(wsp + 12 * MB);   //  2 MiB
  __bf16* wo16 = (__bf16*)(wsp + 14 * MB);   //  2 MiB
  __bf16* qb   = (__bf16*)(wsp + 16 * MB);   //  8 MiB  [B,H,S,64]
  __bf16* kb   = (__bf16*)(wsp + 24 * MB);   //  8 MiB
  __bf16* vb   = (__bf16*)(wsp + 32 * MB);   //  8 MiB
  __bf16* obf  = (__bf16*)(wsp + 40 * MB);   //  8 MiB  attention out bf16

  cvt_kernel<<<512, 256, 0, stream>>>(x,  xb,  BB * SEQ * DIMV);
  cvt_kernel<<<256, 256, 0, stream>>>(wq, wqb, DIMV * DIMV);
  cvt_kernel<<<256, 256, 0, stream>>>(wk, wkb, DIMV * DIMV);
  cvt_kernel<<<256, 256, 0, stream>>>(wv, wvb, DIMV * DIMV);
  cvt_kernel<<<256, 256, 0, stream>>>(wo, wo16, DIMV * DIMV);

  qkv_rope_kernel<<<dim3(BB * SEQ / 128, NH, 3), 256, 0, stream>>>(
      xb, wqb, wkb, wvb, qb, kb, vb);

  attn_kernel<<<dim3(SEQ / 128, NH, BB), 256, 0, stream>>>(qb, kb, vb, obf);

  oproj_kernel<<<dim3(BB * SEQ / 128, DIMV / 64), 256, 0, stream>>>(
      obf, wo16, wob, (float*)d_out);
}
